// MTRL_AttModel_71167608094872
// MI455X (gfx1250) — matrix operand order repaired and checked by reading, not compiled, not run on hardware
//
#include <hip/hip_runtime.h>
#include <cstdint>
#include <cstddef>

typedef __bf16 bf16_t;
typedef __attribute__((ext_vector_type(16))) __bf16 v16bf;
typedef __attribute__((ext_vector_type(8)))  __bf16 v8bf;
typedef __attribute__((ext_vector_type(8)))  float   v8f;

namespace {
constexpr int   kB   = 32;
constexpr int   kN   = 2048;
constexpr int   kH   = 128;
constexpr float kNeg = 9000000000000000.0f;

union V16U { v16bf v; v8bf h[2]; };

__device__ __forceinline__ v8f wmma_bf16(v16bf a, v16bf b, v8f c) {
  return __builtin_amdgcn_wmma_f32_16x16x32_bf16(false, a, false, b, (short)0, c,
                                                 false, false);
}

// Branch-free tanh: 1 - 2/(exp(2x)+1).
// __expf -> v_exp_f32, rcp -> v_rcp_f32 (TRANS ops, co-execute with WMMA).
// Saturates correctly: exp->+inf gives 1, exp->0 gives -1. No EXEC divergence.
__device__ __forceinline__ float tanh_fast(float x) {
  const float e = __expf(2.0f * x);
  return 1.0f - 2.0f * __builtin_amdgcn_rcpf(e + 1.0f);
}
} // namespace

// ---------------------------------------------------------------------------
// Phase 1: q/k/v = tanh(x @ W + b). One wave per 16-token tile, all 3 projs.
// q,k stored row-major bf16 [B*N, H]; v stored transposed bf16 [B, H, N].
// ---------------------------------------------------------------------------
__global__ void __launch_bounds__(256)
mtrl_qkv_proj(const float* __restrict__ x,
              const float* __restrict__ Wq, const float* __restrict__ bq,
              const float* __restrict__ Wk, const float* __restrict__ bk,
              const float* __restrict__ Wv, const float* __restrict__ bv,
              bf16_t* __restrict__ qo, bf16_t* __restrict__ ko,
              bf16_t* __restrict__ vto)
{
  const int wave   = (int)(blockIdx.x * (blockDim.x >> 5) + (threadIdx.x >> 5));
  const int lane   = (int)(threadIdx.x & 31);
  const int row    = lane & 15;
  const int hs     = lane >> 4;
  const int koffA  = hs * 16;   // A-fragment K base (contiguous, matches B)
  const int kbaseB = hs * 16;   // B-fragment K base   (contiguous, matches A)
  const int colB   = lane & 15;
  const int rbase  = hs * 8;    // C/D row base

  const int tok0  = wave * 16;        // 16-token tile (never crosses a batch)
  const int batch = tok0 / kN;
  const int iloc0 = tok0 % kN;

  // A fragments from x (f32 -> bf16 on the fly)
  v16bf a[4];
  const float* xrow = x + (size_t)(tok0 + row) * kH;
#pragma unroll
  for (int kk = 0; kk < 4; ++kk) {
#pragma unroll
    for (int h = 0; h < 16; ++h) {
      const int K = kk * 32 + koffA + h;
      a[kk][h] = (bf16_t)xrow[K];
    }
  }

#pragma unroll
  for (int p = 0; p < 3; ++p) {
    const float* W  = (p == 0) ? Wq : (p == 1) ? Wk : Wv;
    const float* bb = (p == 0) ? bq : (p == 1) ? bk : bv;
#pragma unroll
    for (int nt = 0; nt < 8; ++nt) {
      const int n = nt * 16 + colB;
      v8f c = (v8f){};
#pragma unroll
      for (int kk = 0; kk < 4; ++kk) {
        v16bf bw;
#pragma unroll
        for (int h = 0; h < 16; ++h) {
          const int K = kk * 32 + kbaseB + h;
          bw[h] = (bf16_t)W[(size_t)K * kH + n];
        }
        c = wmma_bf16(a[kk], bw, c);
      }
      const float bias = bb[n];
      if (p == 2) {
        // v: store transposed vT[b][n][token], 8 contiguous bf16 per lane
        bf16_t tmp[8];
#pragma unroll
        for (int r = 0; r < 8; ++r) tmp[r] = (bf16_t)tanh_fast(c[r] + bias);
        uint4 u;
        __builtin_memcpy(&u, tmp, sizeof(u));
        *(uint4*)(vto + ((size_t)batch * kH + n) * kN + iloc0 + rbase) = u;
      } else {
        bf16_t* o = (p == 0) ? qo : ko;
#pragma unroll
        for (int r = 0; r < 8; ++r)
          o[(size_t)(tok0 + rbase + r) * kH + n] = (bf16_t)tanh_fast(c[r] + bias);
      }
    }
  }
}

// ---------------------------------------------------------------------------
// Phase 2: flash attention. One wave per (batch, 16-row) tile; loop over
// 32-column K/V chunks with online softmax. Mask read exactly once (HBM
// bound stream, ~536 MB); K/V re-reads hit L2 (512 KB/batch each).
// ---------------------------------------------------------------------------
__global__ void __launch_bounds__(256)
mtrl_flash_attn(const bf16_t* __restrict__ q, const bf16_t* __restrict__ k,
                const bf16_t* __restrict__ vt, const float* __restrict__ mask,
                float* __restrict__ out)
{
  __shared__ bf16_t pbuf[8][16 * 32];   // per-wave P transpose staging (1 KB each)

  const int wv    = (int)(threadIdx.x >> 5);
  const int lane  = (int)(threadIdx.x & 31);
  const int tile  = (int)(blockIdx.x * 8 + wv);   // 0..4095
  const int batch = tile >> 7;                    // 128 tiles per batch
  const int i0    = (tile & 127) << 4;

  const int row    = lane & 15;
  const int hs     = lane >> 4;
  const int koffA  = hs * 16;
  const int kbaseB = hs * 16;
  const int colB   = lane & 15;
  const int rbase  = hs * 8;

  // persistent q A-fragments (16 rows x 128 K) via 32-byte vector loads
  v16bf aq[4];
  const bf16_t* qrow = q + ((size_t)batch * kN + i0 + row) * kH;
#pragma unroll
  for (int kk = 0; kk < 4; ++kk) {
    V16U t;
    t.h[0] = *(const v8bf*)(qrow + kk * 32 + koffA);
    t.h[1] = *(const v8bf*)(qrow + kk * 32 + koffA + 8);
    aq[kk] = t.v;
  }

  v8f O[8];
#pragma unroll
  for (int nt = 0; nt < 8; ++nt) O[nt] = (v8f){};
  float m[8], l[8];
#pragma unroll
  for (int r = 0; r < 8; ++r) { m[r] = -3.0e38f; l[r] = 0.0f; }

  const bf16_t* kb = k  + (size_t)batch * kN * kH;
  const bf16_t* vb = vt + (size_t)batch * kH * kN;
  const float*  mb = mask + (size_t)batch * kN * kN;

  for (int j0 = 0; j0 < kN; j0 += 32) {
    // ---- scores: 16x32 chunk as two 16x16 WMMA tiles, K=128 in 4 steps ----
    v8f S[2];
#pragma unroll
    for (int t = 0; t < 2; ++t) {
      S[t] = (v8f){};
      const bf16_t* krow = kb + (size_t)(j0 + t * 16 + colB) * kH + kbaseB;
#pragma unroll
      for (int kk = 0; kk < 4; ++kk) {
        const v16bf bk_ = *(const v16bf*)(krow + kk * 32);
        S[t] = wmma_bf16(aq[kk], bk_, S[t]);
      }
    }

    // ---- mask + online softmax (half-wave holds a full score row) ----
    const float* mrow0 = mb + (size_t)(i0 + rbase) * kN + j0 + colB;
    __builtin_prefetch(mrow0 + 32, 0, 1);   // next mask chunk -> global_prefetch
#pragma unroll
    for (int r = 0; r < 8; ++r) {
      const float mv0 = mrow0[(size_t)r * kN];
      const float mv1 = mrow0[(size_t)r * kN + 16];
      S[0][r] = S[0][r] * mv0 - kNeg * (1.0f - mv0);
      S[1][r] = S[1][r] * mv1 - kNeg * (1.0f - mv1);
      float cmax = fmaxf(S[0][r], S[1][r]);
#pragma unroll
      for (int d = 1; d < 16; d <<= 1) cmax = fmaxf(cmax, __shfl_xor(cmax, d, 32));
      const float mn    = fmaxf(m[r], cmax);
      const float alpha = __expf(m[r] - mn);
      m[r] = mn;
      const float p0 = __expf(S[0][r] - mn);
      const float p1 = __expf(S[1][r] - mn);
      S[0][r] = p0; S[1][r] = p1;
      float rs = p0 + p1;
#pragma unroll
      for (int d = 1; d < 16; d <<= 1) rs += __shfl_xor(rs, d, 32);
      l[r] = l[r] * alpha + rs;
#pragma unroll
      for (int nt = 0; nt < 8; ++nt) O[nt][r] *= alpha;
    }

    // ---- P: D layout -> A layout via per-wave LDS slice ----
#pragma unroll
    for (int t = 0; t < 2; ++t)
#pragma unroll
      for (int r = 0; r < 8; ++r)
        pbuf[wv][(rbase + r) * 32 + t * 16 + colB] = (bf16_t)S[t][r];
    __asm__ volatile("s_wait_dscnt 0x0" ::: "memory");

    V16U tp;
    tp.h[0] = *(const v8bf*)(&pbuf[wv][row * 32 + koffA]);
    tp.h[1] = *(const v8bf*)(&pbuf[wv][row * 32 + koffA + 8]);
    const v16bf ap = tp.v;

    // ---- O += P @ V (vT gives contiguous B-fragments) ----
#pragma unroll
    for (int nt = 0; nt < 8; ++nt) {
      const v16bf bv_ =
          *(const v16bf*)(vb + (size_t)(nt * 16 + colB) * kN + j0 + kbaseB);
      O[nt] = wmma_bf16(ap, bv_, O[nt]);
    }
  }

  // ---- epilogue: normalize and store f32 ----
  float* orow = out + ((size_t)batch * kN + i0 + rbase) * kH;
#pragma unroll
  for (int r = 0; r < 8; ++r) {
    const float inv = __builtin_amdgcn_rcpf(l[r]);
#pragma unroll
    for (int nt = 0; nt < 8; ++nt)
      orow[(size_t)r * kH + nt * 16 + colB] = O[nt][r] * inv;
  }
}

// ---------------------------------------------------------------------------
extern "C" void kernel_launch(void* const* d_in, const int* in_sizes, int n_in,
                              void* d_out, int out_size, void* d_ws, size_t ws_size,
                              hipStream_t stream) {
  (void)in_sizes; (void)n_in; (void)out_size; (void)ws_size;
  const float* x    = (const float*)d_in[0];
  const float* mask = (const float*)d_in[1];
  const float* Wv   = (const float*)d_in[2];
  const float* bv   = (const float*)d_in[3];
  const float* Wk   = (const float*)d_in[4];
  const float* bk   = (const float*)d_in[5];
  const float* Wq   = (const float*)d_in[6];
  const float* bq   = (const float*)d_in[7];
  float* out = (float*)d_out;

  const size_t elems = (size_t)kB * kN * kH;        // 8,388,608
  bf16_t* qb  = (bf16_t*)d_ws;                      // 16 MB
  bf16_t* kbf = qb + elems;                         // 16 MB
  bf16_t* vtb = kbf + elems;                        // 16 MB (48 MB total)

  dim3 blk(256);
  dim3 grid_proj((kB * kN / 16) / 8);               // 512 blocks, 4096 waves
  mtrl_qkv_proj<<<grid_proj, blk, 0, stream>>>(x, Wq, bq, Wk, bk, Wv, bv,
                                               qb, kbf, vtb);
  dim3 grid_attn((kB * (kN / 16)) / 8);             // 512 blocks, 4096 waves
  mtrl_flash_attn<<<grid_attn, blk, 0, stream>>>(qb, kbf, vtb, mask, out);
}
